// HMP_33337536151847
// MI455X (gfx1250) — matrix operand order repaired and checked by reading, not compiled, not run on hardware
//
#include <hip/hip_runtime.h>
#include <math.h>

// ---------------------------------------------------------------------------
// MI455X / gfx1250 implementation.
// 3.3M independent tiny MLPs ([1xK]@[Kx32] -> tanh -> [1x32]@[32x32]) mapped
// onto v_wmma_f32_16x16x32_f16 (16-node tiles per wave32). Weights are
// pre-swizzled in LDS into per-lane WMMA fragment order so every B fragment
// is two ds_load_b128. Scatters use f32 atomics that resolve in the 192MB L2
// (largest target = 128MB).
// ---------------------------------------------------------------------------

typedef __attribute__((ext_vector_type(16))) _Float16 v16h;
typedef __attribute__((ext_vector_type(8)))  _Float16 v8h;
typedef __attribute__((ext_vector_type(4)))  _Float16 v4h;
typedef __attribute__((ext_vector_type(8)))  float    v8f;

#define FEAT 32   // U in the reference

__device__ __forceinline__ v8f wmma16x16x32(v16h a, v16h b, v8f c) {
  // 8 args: (neg_a, A, neg_b, B, c_mod, C, reuse_a, reuse_b)
  return __builtin_amdgcn_wmma_f32_16x16x32_f16(
      false, a, false, b, (short)0, c, false, false);
}

// Branchless tanh on the hardware transcendental pipe (v_exp_f32 + v_rcp_f32)
__device__ __forceinline__ float fast_tanh(float x) {
  x = fminf(15.f, fmaxf(-15.f, x));
  const float t = __expf(2.f * x);
  return (t - 1.f) * __builtin_amdgcn_rcpf(t + 1.f);
}

// Concatenate two contiguous 8-half LDS runs into a v16h fragment.
__device__ __forceinline__ v16h load_frag2(const _Float16* lo, const _Float16* hi) {
  const v8h a = *(const v8h*)lo;
  const v8h b = *(const v8h*)hi;
  v16h r;
  #pragma unroll
  for (int i = 0; i < 8; ++i) { r[i] = a[i]; r[8 + i] = b[i]; }
  return r;
}

// Copy one 32-float row to LDS as f16 using float4 loads + packed cvt + b64 LDS stores.
__device__ __forceinline__ void stage_row(const float* __restrict__ src,
                                          _Float16* __restrict__ dst) {
  #pragma unroll
  for (int q = 0; q < 8; ++q) {
    const float4 f = ((const float4*)src)[q];
    v4h o;
    o[0] = (_Float16)f.x; o[1] = (_Float16)f.y;
    o[2] = (_Float16)f.z; o[3] = (_Float16)f.w;
    ((v4h*)dst)[q] = o;
  }
}

// ---------------------------------------------------------------------------
// Fused 2-layer MLP over 16-node tiles.
// KT = number of 32-wide K chunks in layer-1 input (3 = up pass, 2 = down/ring)
//   KT==3: A row = [xA[node] | xB[idx0[node]] | xC[idx1[node]]]
//   KT==2: A row = [xA[node] | xB[node]]               (idx0/idx1 unused)
// Layer1: [16 x 32*KT] @ W1[32*KT x 32] + b1 -> tanh
// Layer2: [16 x 32]    @ W2[32 x 32]    + b2 -> out[node]
// 256 threads = 8 waves, one 16-node tile per wave. All N are multiples of 16.
//
// A-matrix 16x32 f16 layout used here (k = 16*hi + s): per lane, two contiguous
// 8-half runs at k = kb + 16*hi and k = kb + 16*hi + 8  (hi = lane>=16).
// B-matrix 32x16 f16 layout: lane L holds K = kb + 16*hi + i, N = L&15 —
// pre-swizzled into sW1f/sW2f so each lane's 16 halves are contiguous.
// C/D layout: VGPR r, lane L -> element [M = r + 8*hi][N = (L&15) + 16*nt].
// ---------------------------------------------------------------------------
template <int KT>
__global__ __launch_bounds__(256)
void mlp_wmma(const float* __restrict__ xA,
              const float* __restrict__ xB,
              const float* __restrict__ xC,
              const int*   __restrict__ idx0,
              const int*   __restrict__ idx1,
              const float* __restrict__ W1, const float* __restrict__ b1,
              const float* __restrict__ W2, const float* __restrict__ b2,
              float* __restrict__ out, int ntiles)
{
  constexpr int K1 = KT * 32;

  // weights in per-lane fragment order: [frag][lane][16 halves]
  __shared__ __align__(32) _Float16 sW1f[KT * 2 * 32 * 16];
  __shared__ __align__(32) _Float16 sW2f[2 * 32 * 16];
  __shared__ float    sB1[32], sB2[32];
  __shared__ __align__(32) _Float16 sA[8][16 * K1];   // per-wave A tile [16][K1]
  __shared__ __align__(32) _Float16 sX[8][16 * 32];   // per-wave layer-1 out

  const int tid = threadIdx.x;

  // ---- stage + swizzle weights once per block (tiny, L2-resident) --------
  // fragment (kc, nt), lane L, half i  <-  W1[(kc*32 + 16*(L>>4) + i)*32 + (L&15) + 16*nt]
  for (int t = tid; t < K1 * 32; t += 256) {
    const int frag = t >> 9, within = t & 511;
    const int L = within >> 4, i = within & 15;
    const int kc = frag >> 1, nt = frag & 1;
    sW1f[t] = (_Float16)W1[(kc * 32 + ((L >> 4) << 4) + i) * 32 + (L & 15) + (nt << 4)];
  }
  for (int t = tid; t < 2 * 512; t += 256) {
    const int nt = t >> 9, within = t & 511;
    const int L = within >> 4, i = within & 15;
    sW2f[t] = (_Float16)W2[(((L >> 4) << 4) + i) * 32 + (L & 15) + (nt << 4)];
  }
  if (tid < 32) { sB1[tid] = b1[tid]; sB2[tid] = b2[tid]; }
  __syncthreads();

  const int w    = tid >> 5;   // wave id in block
  const int lane = tid & 31;
  const int tile = blockIdx.x * 8 + w;
  if (tile >= ntiles) return;  // no workgroup barriers past this point
  const int base = tile * 16;  // first node of this tile

  // ---- stage gathered activations [16 x K1] f32 -> f16 into LDS ----------
  if constexpr (KT == 3) {
    // 48 row-copies (16 nodes x 3 sources) over 32 lanes
    for (int t = lane; t < 48; t += 32) {
      const int m = t / 3, s = t - 3 * m;
      const int node = base + m;
      const float* src =
          (s == 0) ? (xA + (size_t)node * FEAT)
        : (s == 1) ? (xB + (size_t)idx0[node] * FEAT)
                   : (xC + (size_t)idx1[node] * FEAT);
      stage_row(src, &sA[w][m * K1 + s * FEAT]);
    }
  } else {
    // 32 row-copies (16 nodes x 2 sources), one per lane
    const int m = lane >> 1, s = lane & 1;
    const int node = base + m;
    const float* src = (s == 0) ? (xA + (size_t)node * FEAT)
                                : (xB + (size_t)node * FEAT);
    stage_row(src, &sA[w][m * K1 + s * FEAT]);
  }
  // wave-private LDS: DS ops are in-order within a wave; just stop the
  // compiler from hoisting the loads above the stores.
  __builtin_amdgcn_wave_barrier();

  const int row = lane & 15;
  const int hi  = lane >> 4;   // 0: lanes 0-15, 1: lanes 16-31

  // ---- layer 1: A[16xK1] @ W1[K1x32] -> +b1 -> tanh -> sX ----------------
  #pragma unroll
  for (int nt = 0; nt < 2; ++nt) {
    const int n = row + nt * 16;           // output column for C/D layout
    v8f c = {};
    #pragma unroll
    for (int kc = 0; kc < KT; ++kc) {
      const int kb = kc * 32;
      const v16h a = load_frag2(&sA[w][row * K1 + kb + hi * 16],
                                &sA[w][row * K1 + kb + 8 + hi * 16]);
      const v16h bf = *(const v16h*)&sW1f[((kc * 2 + nt) * 32 + lane) * 16];
      c = wmma16x16x32(a, bf, c);
    }
    #pragma unroll
    for (int r = 0; r < 8; ++r) {
      const int m = r + hi * 8;
      sX[w][m * 32 + n] = (_Float16)fast_tanh(c[r] + sB1[n]);
    }
  }
  __builtin_amdgcn_wave_barrier();

  // ---- layer 2: sX[16x32] @ W2[32x32] -> +b2 -> out ----------------------
  #pragma unroll
  for (int nt = 0; nt < 2; ++nt) {
    const int n = row + nt * 16;
    const v16h a = load_frag2(&sX[w][row * 32 + hi * 16],
                              &sX[w][row * 32 + 8 + hi * 16]);
    const v16h bf = *(const v16h*)&sW2f[(nt * 32 + lane) * 16];
    v8f c = {};
    c = wmma16x16x32(a, bf, c);
    #pragma unroll
    for (int r = 0; r < 8; ++r) {
      const int m = r + hi * 8;
      out[(size_t)(base + m) * FEAT + n] = c[r] + sB2[n];
    }
  }
}

// ---------------------------------------------------------------------------
// dst[i0[e]] += src[e]; dst[i1[e]] += src[e]   (32-float rows, 1 thread/elem)
// Lanes of a wave cover one row's 32 contiguous floats -> coalesced atomics
// that resolve in the 192MB L2 (largest dst region = 128MB).
// ---------------------------------------------------------------------------
__global__ __launch_bounds__(256)
void scatter_add2(const float* __restrict__ src,
                  const int* __restrict__ i0, const int* __restrict__ i1,
                  float* __restrict__ dst, int nE)
{
  const size_t idx = (size_t)blockIdx.x * blockDim.x + threadIdx.x;
  if (idx >= (size_t)nE * FEAT) return;
  const int e = (int)(idx >> 5), f = (int)(idx & 31);
  const float v = src[idx];
  atomicAdd(&dst[(size_t)i0[e] * FEAT + f], v);
  atomicAdd(&dst[(size_t)i1[e] * FEAT + f], v);
}

// dst[sidx[e]] += src[gidx[e]]  (ring segment-sum and ring broadcast-back)
__global__ __launch_bounds__(256)
void gather_scatter_add(const float* __restrict__ src,
                        const int* __restrict__ gidx,
                        const int* __restrict__ sidx,
                        float* __restrict__ dst, int nE)
{
  const size_t idx = (size_t)blockIdx.x * blockDim.x + threadIdx.x;
  if (idx >= (size_t)nE * FEAT) return;
  const int e = (int)(idx >> 5), f = (int)(idx & 31);
  const float v = src[(size_t)gidx[e] * FEAT + f];
  atomicAdd(&dst[(size_t)sidx[e] * FEAT + f], v);
}

// ---------------------------------------------------------------------------

extern "C" void kernel_launch(void* const* d_in, const int* in_sizes, int n_in,
                              void* d_out, int out_size, void* d_ws, size_t ws_size,
                              hipStream_t stream)
{
  const float* h1 = (const float*)d_in[0];
  const float* h2 = (const float*)d_in[1];
  const float* h3 = (const float*)d_in[2];
  const float* h4 = (const float*)d_in[3];
  const float* upW1   = (const float*)d_in[4];
  const float* upb1   = (const float*)d_in[5];
  const float* upW2   = (const float*)d_in[6];
  const float* upb2   = (const float*)d_in[7];
  const float* downW1 = (const float*)d_in[8];
  const float* downb1 = (const float*)d_in[9];
  const float* downW2 = (const float*)d_in[10];
  const float* downb2 = (const float*)d_in[11];
  const float* ringW1 = (const float*)d_in[12];
  const float* ringb1 = (const float*)d_in[13];
  const float* ringW2 = (const float*)d_in[14];
  const float* ringb2 = (const float*)d_in[15];
  const int* idx2_0 = (const int*)d_in[16];
  const int* idx2_1 = (const int*)d_in[17];
  const int* idx3_0 = (const int*)d_in[18];
  const int* idx3_1 = (const int*)d_in[19];
  const int* idx4_0 = (const int*)d_in[20];
  const int* idx4_1 = (const int*)d_in[21];
  const int* ring_src = (const int*)d_in[22];
  const int* ring_dst = (const int*)d_in[23];

  const int N1 = in_sizes[0] / FEAT;
  const int N2 = in_sizes[1] / FEAT;
  const int N3 = in_sizes[2] / FEAT;
  const int N4 = in_sizes[3] / FEAT;
  const int ER = in_sizes[22];
  const int NR = out_size / FEAT - (N1 + N2 + N3 + N4);  // n_ring from shapes
  (void)NR; (void)n_in; (void)ws_size;

  float* out_h1   = (float*)d_out;
  float* out_h2   = out_h1 + (size_t)N1 * FEAT;
  float* out_h3   = out_h2 + (size_t)N2 * FEAT;
  float* out_h4   = out_h3 + (size_t)N3 * FEAT;
  float* out_ring = out_h4 + (size_t)N4 * FEAT;

  float* acc = (float*)d_ws;   // reused scatter accumulator (<= N3*32 f32)

  auto mlp_blocks  = [](int n) { return (n / 16 + 7) / 8; };
  auto elem_blocks = [](int n) { return (int)(((size_t)n * FEAT + 255) / 256); };

  // ---- up pass: k = 2,3,4  (h4 result is final) --------------------------
  mlp_wmma<3><<<mlp_blocks(N2), 256, 0, stream>>>(
      h2, h1, h1, idx2_0, idx2_1,
      upW1 + 0 * 96 * 32, upb1 + 0 * 32, upW2 + 0 * 32 * 32, upb2 + 0 * 32,
      out_h2, N2 / 16);
  mlp_wmma<3><<<mlp_blocks(N3), 256, 0, stream>>>(
      h3, out_h2, out_h2, idx3_0, idx3_1,
      upW1 + 1 * 96 * 32, upb1 + 1 * 32, upW2 + 1 * 32 * 32, upb2 + 1 * 32,
      out_h3, N3 / 16);
  mlp_wmma<3><<<mlp_blocks(N4), 256, 0, stream>>>(
      h4, out_h3, out_h3, idx4_0, idx4_1,
      upW1 + 2 * 96 * 32, upb1 + 2 * 32, upW2 + 2 * 32 * 32, upb2 + 2 * 32,
      out_h4, N4 / 16);

  // ---- down pass: k = 4 -> h3, k = 3 -> h2, k = 2 -> h1 ------------------
  hipMemsetAsync(acc, 0, (size_t)N3 * FEAT * sizeof(float), stream);
  scatter_add2<<<elem_blocks(N4), 256, 0, stream>>>(out_h4, idx4_0, idx4_1, acc, N4);
  mlp_wmma<2><<<mlp_blocks(N3), 256, 0, stream>>>(
      out_h3, acc, nullptr, nullptr, nullptr,
      downW1 + 2 * 64 * 32, downb1 + 2 * 32, downW2 + 2 * 32 * 32, downb2 + 2 * 32,
      out_h3, N3 / 16);   // in-place: each tile touches only its own rows

  hipMemsetAsync(acc, 0, (size_t)N2 * FEAT * sizeof(float), stream);
  scatter_add2<<<elem_blocks(N3), 256, 0, stream>>>(out_h3, idx3_0, idx3_1, acc, N3);
  mlp_wmma<2><<<mlp_blocks(N2), 256, 0, stream>>>(
      out_h2, acc, nullptr, nullptr, nullptr,
      downW1 + 1 * 64 * 32, downb1 + 1 * 32, downW2 + 1 * 32 * 32, downb2 + 1 * 32,
      out_h2, N2 / 16);

  hipMemsetAsync(acc, 0, (size_t)N1 * FEAT * sizeof(float), stream);
  scatter_add2<<<elem_blocks(N2), 256, 0, stream>>>(out_h2, idx2_0, idx2_1, acc, N2);
  mlp_wmma<2><<<mlp_blocks(N1), 256, 0, stream>>>(
      h1, acc, nullptr, nullptr, nullptr,
      downW1 + 0 * 64 * 32, downb1 + 0 * 32, downW2 + 0 * 32 * 32, downb2 + 0 * 32,
      out_h1, N1 / 16);

  // ---- ring: segment-sum h1''[src] by dst, broadcast back, final MLP -----
  hipMemsetAsync(out_ring, 0, (size_t)NR * FEAT * sizeof(float), stream);
  gather_scatter_add<<<elem_blocks(ER), 256, 0, stream>>>(
      out_h1, ring_src, ring_dst, out_ring, ER);          // h_ring (final output)

  hipMemsetAsync(acc, 0, (size_t)N1 * FEAT * sizeof(float), stream);
  gather_scatter_add<<<elem_blocks(ER), 256, 0, stream>>>(
      out_ring, ring_dst, ring_src, acc, ER);             // h_ring_down

  mlp_wmma<2><<<mlp_blocks(N1), 256, 0, stream>>>(
      out_h1, acc, nullptr, nullptr, nullptr,
      ringW1, ringb1, ringW2, ringb2,
      out_h1, N1 / 16);
}
